// Transformer_9363028705772
// MI455X (gfx1250) — compile-verified, matrix operand order checked by reading, not run on hardware
//
#include <hip/hip_runtime.h>
#include <math.h>

// ---------------------------------------------------------------- constants
#define LAYERS 4
#define DIM    1024
#define FDIM   4096
#define HEADS  16
#define DH     64
#define NIN    128
#define SLEN   129          // S = NIN + 1
#define BSZ    256
#define TOK    (BSZ * SLEN) // 33024 tokens, divisible by 64
#define EPS    1e-5f
#define NEGBIG 1e10f

// WMMA types (gfx1250, wave32)
typedef __attribute__((ext_vector_type(16))) __bf16 v16bf;
typedef __attribute__((ext_vector_type(8)))  float  v8f;

// f32 -> bf16 with round-to-nearest-even
__device__ __forceinline__ __bf16 f2bf(float f) {
    union { float f; unsigned u; } a; a.f = f;
    unsigned r = a.u + 0x7FFFu + ((a.u >> 16) & 1u);
    union { unsigned short s; __bf16 b; } o; o.s = (unsigned short)(r >> 16);
    return o.b;
}

// ISA 16-bit A/B fragment placement (CDNA5 ISA §7.12.2):
// lanes 0-15 hold K = 0..7,16..23 ; lanes 16-31 hold K = 8..15,24..31.
// Element (rn in 0..15, k in 0..31) -> lane = rn + 16*((k>>3)&1),
// elem e = (k&7) + 8*(k>>4).  Each lane's 16 bf16 are contiguous (32B).
__device__ __forceinline__ int frag_off(int rn, int k) {
    int lane = rn + (((k >> 3) & 1) << 4);
    int e    = (k & 7) | ((k >> 4) << 3);
    return lane * 16 + e;
}

// ---- CDNA5 async global->LDS DMA (ASYNCcnt tracked, §15.18.3 op 98) ------
__device__ __forceinline__ void async_copy16(unsigned lds_off, const void* g) {
    asm volatile("global_load_async_to_lds_b128 %0, %1, off"
                 :: "v"(lds_off), "v"(g) : "memory");
}
__device__ __forceinline__ void wait_async_le4() {
    asm volatile("s_wait_asynccnt 0x4" ::: "memory");
}
__device__ __forceinline__ void wait_async_0() {
    asm volatile("s_wait_asynccnt 0x0" ::: "memory");
}

// ------------------------------------------------------------- embed kernel
__global__ void embed_kernel(const int* __restrict__ x,
                             const float* __restrict__ emb,
                             const float* __restrict__ pos,
                             float* __restrict__ h) {
    size_t i = (size_t)blockIdx.x * blockDim.x + threadIdx.x;
    size_t total = (size_t)TOK * DIM;
    if (i >= total) return;
    int d = (int)(i & (DIM - 1));
    size_t tok = i >> 10;
    int s = (int)(tok % SLEN);
    int b = (int)(tok / SLEN);
    float v = pos[s * DIM + d];
    if (s > 0) v += emb[(size_t)x[b * NIN + (s - 1)] * DIM + d];
    h[i] = v;
}

// -------------------------------------------- weight conversion + tiling
// f32 row-major [K][N]  ->  bf16 fragment-tiled [K/32][N/16][512]
__global__ void cvt_tileB(const float* __restrict__ in,
                          __bf16* __restrict__ out, int K, int N) {
    size_t i = (size_t)blockIdx.x * blockDim.x + threadIdx.x;
    if (i >= (size_t)K * N) return;
    int k = (int)(i / N);
    int n = (int)(i - (size_t)k * N);
    size_t tile = (size_t)(k >> 5) * (N >> 4) + (n >> 4);
    out[tile * 512 + frag_off(n & 15, k & 31)] = f2bf(in[i]);
}

// --------------------------------------------------------------- layernorm
__device__ __forceinline__ void stval(float* p, float v)  { *p = v; }
__device__ __forceinline__ void stval(__bf16* p, float v) { *p = f2bf(v); }

template <typename OUT>
__global__ void ln_kernel(const float* __restrict__ in,
                          const float* __restrict__ sc,
                          const float* __restrict__ sh,
                          OUT* __restrict__ out) {
    int row = blockIdx.x;
    int tid = threadIdx.x;
    const float* p = in + (size_t)row * DIM;
    float vals[4], s = 0.f, s2 = 0.f;
#pragma unroll
    for (int i = 0; i < 4; ++i) {
        float v = p[tid + i * 256];
        vals[i] = v; s += v; s2 += v * v;
    }
    __shared__ float r1[256], r2[256];
    r1[tid] = s; r2[tid] = s2;
    __syncthreads();
    for (int st = 128; st > 0; st >>= 1) {
        if (tid < st) { r1[tid] += r1[tid + st]; r2[tid] += r2[tid + st]; }
        __syncthreads();
    }
    float mean = r1[0] * (1.0f / DIM);
    float var  = r2[0] * (1.0f / DIM) - mean * mean;
    float inv  = rsqrtf(var + EPS);
    OUT* o = out + (size_t)row * DIM;
#pragma unroll
    for (int i = 0; i < 4; ++i) {
        int d = tid + i * 256;
        stval(o + d, sc[d] * ((vals[i] - mean) * inv) + sh[d]);
    }
}

// ------------------------------------------------------ bf16 WMMA GEMM
// C[M,N] = A[M,K](bf16 row-major) @ Bt(bf16 fragment-tiled) + bias
// (+residual f32) (relu?)  -> Cf (f32) or Cbf (bf16 row-major)
// block tile 64x64, 128 thr = 4 wave32, wave -> 32x32 (2x2 WMMA frags).
// Tiles staged via double-buffered GLOBAL_LOAD_ASYNC_TO_LDS_B128.
#define BM 64
#define BN 64
#define BK 32
__global__ __launch_bounds__(128)
void gemm_bf16_wmma(const __bf16* __restrict__ A,
                    const __bf16* __restrict__ Bt,
                    const float*  __restrict__ bias,
                    const float*  __restrict__ residual,
                    float*  __restrict__ Cf,
                    __bf16* __restrict__ Cbf,
                    int N, int K, int relu) {
    // [2 buffers][ A: 4KB | B: 4KB ]
    __shared__ __align__(16) char lds[2 * 8192];

    const int tid  = threadIdx.x;
    const int lane = tid & 31;
    const int wave = tid >> 5;
    const int wm   = wave & 1;
    const int wn   = wave >> 1;
    const int rowBase = blockIdx.y * BM;
    const int colBase = blockIdx.x * BN;
    const unsigned ldsbase = (unsigned)(size_t)&lds[0]; // low 32b = LDS offset

    // ---- per-thread async chunk setup (4 x 16B chunks per slab) ----------
    // A tile 64x32 bf16 = 256 chunks; chunk i: mrow=i>>2, c=i&3 covers
    // k = 8c..8c+7 of row mrow -> contiguous 16B in fragment layout.
    const int ia0 = tid, ia1 = tid + 128;
    const int am0 = ia0 >> 2, ac0 = ia0 & 3;
    const int am1 = ia1 >> 2, ac1 = ia1 & 3;
    size_t ag0 = (size_t)(rowBase + am0) * K + ac0 * 8;
    size_t ag1 = (size_t)(rowBase + am1) * K + ac1 * 8;
    const unsigned al0 = ((am0 >> 4) << 10) + ((am0 & 15) + ((ac0 & 1) << 4)) * 32 + ((ac0 >> 1) << 4);
    const unsigned al1 = ((am1 >> 4) << 10) + ((am1 & 15) + ((ac1 & 1) << 4)) * 32 + ((ac1 >> 1) << 4);
    // B tile: 4 x 512-elem fragment blocks, contiguous in Bt.
    const int jb0 = tid, jb1 = tid + 128;
    const size_t bslab = (size_t)(N >> 4) * 512;   // per-k0-slab advance
    size_t bg0 = ((size_t)(colBase >> 4) + (jb0 >> 6)) * 512 + (jb0 & 63) * 8;
    size_t bg1 = ((size_t)(colBase >> 4) + (jb1 >> 6)) * 512 + (jb1 & 63) * 8;
    const unsigned bl0 = 4096 + ((jb0 >> 6) << 10) + ((jb0 & 63) << 4);
    const unsigned bl1 = 4096 + ((jb1 >> 6) << 10) + ((jb1 & 63) << 4);

    const int nslab = K >> 5;
    v8f acc[2][2] = {};

    // prologue: slab 0 -> buffer 0
    {
        const unsigned lb = ldsbase;
        async_copy16(lb + al0, A + ag0);
        async_copy16(lb + al1, A + ag1);
        async_copy16(lb + bl0, Bt + bg0);
        async_copy16(lb + bl1, Bt + bg1);
    }

    for (int s = 0; s < nslab; ++s) {
        if (s + 1 < nslab) {
            ag0 += BK; ag1 += BK; bg0 += bslab; bg1 += bslab;
            const unsigned lb = ldsbase + (((s + 1) & 1) ? 8192u : 0u);
            async_copy16(lb + al0, A + ag0);
            async_copy16(lb + al1, A + ag1);
            async_copy16(lb + bl0, Bt + bg0);
            async_copy16(lb + bl1, Bt + bg1);
            wait_async_le4();   // in-order ASYNCcnt: slab s has landed
        } else {
            wait_async_0();
        }
        __syncthreads();

        const char* buf = lds + ((s & 1) ? 8192 : 0);
        v16bf a0 = *(const v16bf*)(buf + (wm * 2 + 0) * 1024 + lane * 32);
        v16bf a1 = *(const v16bf*)(buf + (wm * 2 + 1) * 1024 + lane * 32);
        v16bf b0 = *(const v16bf*)(buf + 4096 + (wn * 2 + 0) * 1024 + lane * 32);
        v16bf b1 = *(const v16bf*)(buf + 4096 + (wn * 2 + 1) * 1024 + lane * 32);

        acc[0][0] = __builtin_amdgcn_wmma_f32_16x16x32_bf16(false, a0, false, b0, (short)0, acc[0][0], false, false);
        acc[0][1] = __builtin_amdgcn_wmma_f32_16x16x32_bf16(false, a0, false, b1, (short)0, acc[0][1], false, false);
        acc[1][0] = __builtin_amdgcn_wmma_f32_16x16x32_bf16(false, a1, false, b0, (short)0, acc[1][0], false, false);
        acc[1][1] = __builtin_amdgcn_wmma_f32_16x16x32_bf16(false, a1, false, b1, (short)0, acc[1][1], false, false);
        __syncthreads();
    }

    // epilogue: C/D layout -> VGPR r, lanes 0-15: M=r, lanes 16-31: M=r+8
    const int lhalf = lane >> 4;
    const int lcol  = lane & 15;
#pragma unroll
    for (int mi = 0; mi < 2; ++mi) {
#pragma unroll
        for (int ni = 0; ni < 2; ++ni) {
            int col  = colBase + wn * 32 + ni * 16 + lcol;
            int row0 = rowBase + wm * 32 + mi * 16 + lhalf * 8;
            float bv = bias[col];
            v8f c = acc[mi][ni];
#pragma unroll
            for (int r = 0; r < 8; ++r) {
                int row = row0 + r;
                float v = c[r] + bv;
                if (residual) v += residual[(size_t)row * N + col];
                if (relu) v = fmaxf(v, 0.f);
                if (Cbf) Cbf[(size_t)row * N + col] = f2bf(v);
                else     Cf [(size_t)row * N + col] = v;
            }
        }
    }
}

// ------------------------------------------------------------- attention
// one 128-thread block per (b, h, s) query row. qkv: [TOK, 3*DIM] f32.
__global__ __launch_bounds__(128)
void attn_kernel(const float* __restrict__ qkv,
                 __bf16* __restrict__ out, int layer) {
    int s  = blockIdx.x % SLEN;
    int bh = blockIdx.x / SLEN;
    int h  = bh & (HEADS - 1);
    int b  = bh >> 4;
    int tid = threadIdx.x;

    __shared__ float qv[DH];
    __shared__ float scv[256];
    __shared__ float red[128];

    size_t qoff = ((size_t)(b * SLEN + s)) * (3 * DIM) + h * DH;
    if (tid < DH) qv[tid] = qkv[qoff + tid];
    __syncthreads();

    for (int t = tid; t < 256; t += 128) {
        float v = -INFINITY;
        if (t < SLEN) {
            const float* kp = qkv + ((size_t)(b * SLEN + t)) * (3 * DIM) + DIM + h * DH;
            float d = 0.f;
#pragma unroll 8
            for (int j = 0; j < DH; ++j) d += qv[j] * kp[j];
            // MADE masks (natural ordering)
            bool mv = (layer == 0)
                    ? ((s < NIN) && (t == 0 || s >= t))
                    : ((s < NIN) && (t < NIN) && (s >= t));
            v = mv ? d * 0.125f : -NEGBIG;
        }
        scv[t] = v;
    }
    __syncthreads();

    red[tid] = fmaxf(scv[tid], scv[tid + 128]);
    __syncthreads();
    for (int st = 64; st > 0; st >>= 1) {
        if (tid < st) red[tid] = fmaxf(red[tid], red[tid + st]);
        __syncthreads();
    }
    float mx = red[0];
    __syncthreads();

    float sum = 0.f;
    for (int t = tid; t < SLEN; t += 128) {
        float e = __expf(scv[t] - mx);
        scv[t] = e;
        sum += e;
    }
    red[tid] = sum;
    __syncthreads();
    for (int st = 64; st > 0; st >>= 1) {
        if (tid < st) red[tid] += red[tid + st];
        __syncthreads();
    }
    float inv = 1.f / red[0];

    if (tid < DH) {
        const float* vp = qkv + ((size_t)(b * SLEN)) * (3 * DIM) + 2 * DIM + h * DH + tid;
        float acc = 0.f;
        for (int t = 0; t < SLEN; ++t) acc += scv[t] * vp[(size_t)t * (3 * DIM)];
        out[((size_t)(b * SLEN + s)) * DIM + h * DH + tid] = f2bf(acc * inv);
    }
}

// ---------------------------------------------------------------- launcher
extern "C" void kernel_launch(void* const* d_in, const int* in_sizes, int n_in,
                              void* d_out, int out_size, void* d_ws, size_t ws_size,
                              hipStream_t stream) {
    const int*   x       = (const int*)  d_in[0];
    const float* embed   = (const float*)d_in[1];
    const float* pos_emb = (const float*)d_in[2];
    const float* qkv_w   = (const float*)d_in[3];
    const float* qkv_b   = (const float*)d_in[4];
    const float* proj_w  = (const float*)d_in[5];
    const float* proj_b  = (const float*)d_in[6];
    const float* ln1_s   = (const float*)d_in[7];
    const float* ln1_b   = (const float*)d_in[8];
    const float* ln2_s   = (const float*)d_in[9];
    const float* ln2_b   = (const float*)d_in[10];
    const float* mlp_w1  = (const float*)d_in[11];
    const float* mlp_b1  = (const float*)d_in[12];
    const float* mlp_w2  = (const float*)d_in[13];
    const float* mlp_b2  = (const float*)d_in[14];
    const float* lnf_s   = (const float*)d_in[15];
    const float* lnf_b   = (const float*)d_in[16];
    float* outp = (float*)d_out;

    // workspace carve (256B aligned regions)
    char* wsp = (char*)d_ws;
    auto carve = [&](size_t bytes) -> void* {
        void* p = (void*)wsp;
        wsp += (bytes + 255) & ~(size_t)255;
        return p;
    };
    float*  hbuf  = (float*) carve((size_t)TOK * DIM * 4);      // residual stream
    float*  obuf  = (float*) carve((size_t)TOK * DIM * 4);      // post-proj
    float*  qkvb  = (float*) carve((size_t)TOK * 3 * DIM * 4);  // qkv (f32, attn)
    __bf16* abuf  = (__bf16*)carve((size_t)TOK * DIM * 2);      // LN out (bf16)
    __bf16* attno = (__bf16*)carve((size_t)TOK * DIM * 2);      // attn out (bf16)
    __bf16* t1    = (__bf16*)carve((size_t)TOK * FDIM * 2);     // mlp hidden (bf16)
    __bf16* wtq   = (__bf16*)carve((size_t)LAYERS * DIM * 3 * DIM * 2);
    __bf16* wtp   = (__bf16*)carve((size_t)LAYERS * DIM * DIM * 2);
    __bf16* wt1   = (__bf16*)carve((size_t)LAYERS * DIM * FDIM * 2);
    __bf16* wt2   = (__bf16*)carve((size_t)LAYERS * FDIM * DIM * 2);

    // one-time weight conversion -> bf16 fragment-tiled layout
    for (int i = 0; i < LAYERS; ++i) {
        size_t nq = (size_t)DIM * 3 * DIM;
        size_t np = (size_t)DIM * DIM;
        size_t n1 = (size_t)DIM * FDIM;
        size_t n2 = (size_t)FDIM * DIM;
        cvt_tileB<<<dim3((unsigned)((nq + 255) / 256)), dim3(256), 0, stream>>>(
            qkv_w + i * nq, wtq + i * nq, DIM, 3 * DIM);
        cvt_tileB<<<dim3((unsigned)((np + 255) / 256)), dim3(256), 0, stream>>>(
            proj_w + i * np, wtp + i * np, DIM, DIM);
        cvt_tileB<<<dim3((unsigned)((n1 + 255) / 256)), dim3(256), 0, stream>>>(
            mlp_w1 + i * n1, wt1 + i * n1, DIM, FDIM);
        cvt_tileB<<<dim3((unsigned)((n2 + 255) / 256)), dim3(256), 0, stream>>>(
            mlp_w2 + i * n2, wt2 + i * n2, FDIM, DIM);
    }

    // embedding + positions
    {
        size_t total = (size_t)TOK * DIM;
        embed_kernel<<<dim3((unsigned)((total + 255) / 256)), dim3(256), 0, stream>>>(
            x, embed, pos_emb, hbuf);
    }

    for (int i = 0; i < LAYERS; ++i) {
        // LN1 -> bf16
        ln_kernel<__bf16><<<dim3(TOK), dim3(256), 0, stream>>>(
            hbuf, ln1_s + i * DIM, ln1_b + i * DIM, abuf);
        // QKV GEMM: [TOK,1024] x [1024,3072] -> f32
        gemm_bf16_wmma<<<dim3(3 * DIM / BN, TOK / BM), dim3(128), 0, stream>>>(
            abuf, wtq + (size_t)i * DIM * 3 * DIM, qkv_b + (size_t)i * 3 * DIM,
            nullptr, qkvb, nullptr, 3 * DIM, DIM, 0);
        // masked attention -> bf16
        attn_kernel<<<dim3(BSZ * HEADS * SLEN), dim3(128), 0, stream>>>(
            qkvb, attno, i);
        // proj GEMM (+ residual, layer > 0) -> f32
        gemm_bf16_wmma<<<dim3(DIM / BN, TOK / BM), dim3(128), 0, stream>>>(
            attno, wtp + (size_t)i * DIM * DIM, proj_b + (size_t)i * DIM,
            (i > 0) ? hbuf : nullptr, obuf, nullptr, DIM, DIM, 0);
        // LN2 -> bf16
        ln_kernel<__bf16><<<dim3(TOK), dim3(256), 0, stream>>>(
            obuf, ln2_s + i * DIM, ln2_b + i * DIM, abuf);
        // MLP up (ReLU) -> bf16
        gemm_bf16_wmma<<<dim3(FDIM / BN, TOK / BM), dim3(128), 0, stream>>>(
            abuf, wt1 + (size_t)i * DIM * FDIM, mlp_b1 + (size_t)i * FDIM,
            nullptr, nullptr, t1, FDIM, DIM, 1);
        // MLP down (+ residual, layer > 0) -> f32
        gemm_bf16_wmma<<<dim3(DIM / BN, TOK / BM), dim3(128), 0, stream>>>(
            t1, wt2 + (size_t)i * FDIM * DIM, mlp_b2 + (size_t)i * DIM,
            (i > 0) ? obuf : nullptr, hbuf, nullptr, DIM, FDIM, 0);
    }

    // final LN -> f32 output
    ln_kernel<float><<<dim3(TOK), dim3(256), 0, stream>>>(hbuf, lnf_s, lnf_b, outp);
}
